// MultiScaleAdaptiveFeatSimLoss_41472204210807
// MI455X (gfx1250) — matrix operand order repaired and checked by reading, not compiled, not run on hardware
//
#include <hip/hip_runtime.h>
#include <hip/hip_bf16.h>

// ---------------------------------------------------------------------------
// MultiScaleAdaptiveFeatSimLoss for MI455X (gfx1250, wave32, WMMA)
//
// Sizes: B=2, C=19, H=W=128, CF=64, HF=WF=64, K=7, DIL=2, SIGMA=30, TOPK=8.
// sim is constant over 2x2 output blocks (upsample factor == dilation), so it
// is computed once at 64x64 via f16 WMMA Gram tiles (16x16x64 per tile).
// ---------------------------------------------------------------------------

typedef __attribute__((ext_vector_type(16))) _Float16 v16h;
typedef __attribute__((ext_vector_type(8)))  float    v8f;

#define NPIX   32768        // B*H*W
#define HWIM   16384        // H*W
#define CHN    19
#define K2     49
#define SIG2   900.0f

// ---------------- block reduction helper (deterministic) -------------------
__device__ __forceinline__ float block_reduce(float v, float* red) {
  int t = threadIdx.x;
  red[t] = v;
  __syncthreads();
  for (int s = blockDim.x >> 1; s > 0; s >>= 1) {
    if (t < s) red[t] += red[t + s];
    __syncthreads();
  }
  float r = red[0];
  __syncthreads();
  return r;
}

// ---------------- 1. channel softmax of logits_trg -------------------------
__global__ __launch_bounds__(256)
void softmax_kernel(const float* __restrict__ logits, float* __restrict__ prob) {
  int i = blockIdx.x * 256 + threadIdx.x;
  if (i >= NPIX) return;
  int b = i >> 14, yx = i & (HWIM - 1);
  const float* L = logits + (size_t)b * CHN * HWIM + yx;
  float v[CHN], m = -1e30f;
  #pragma unroll
  for (int c = 0; c < CHN; ++c) { v[c] = L[c * HWIM]; m = fmaxf(m, v[c]); }
  float s = 0.f;
  #pragma unroll
  for (int c = 0; c < CHN; ++c) { v[c] = __expf(v[c] - m); s += v[c]; }
  float inv = 1.0f / s;
  float* P = prob + (size_t)b * CHN * HWIM + yx;
  #pragma unroll
  for (int c = 0; c < CHN; ++c) P[c * HWIM] = v[c] * inv;
}

// ---------------- 2. cross-prob cpp[b,k,y,x] -------------------------------
__global__ __launch_bounds__(256)
void cpp_kernel(const float* __restrict__ prob, float* __restrict__ cpp) {
  int i = blockIdx.x * 256 + threadIdx.x;
  if (i >= NPIX) return;
  int b = i >> 14, yx = i & (HWIM - 1);
  int y = yx >> 7, x = yx & 127;
  const float* Pb = prob + (size_t)b * CHN * HWIM;
  float pc[CHN];
  #pragma unroll
  for (int c = 0; c < CHN; ++c) pc[c] = Pb[c * HWIM + yx];
  for (int k = 0; k < K2; ++k) {
    int dy = (k / 7 - 3) * 2, dx = (k % 7 - 3) * 2;
    int ny = y + dy, nx = x + dx;
    float s = 0.f;
    if (ny >= 0 && ny < 128 && nx >= 0 && nx < 128) {
      const float* Pn = Pb + ny * 128 + nx;
      #pragma unroll
      for (int c = 0; c < CHN; ++c) s += pc[c] * Pn[c * HWIM];
    }
    cpp[(((size_t)b * K2 + k) << 14) + yx] = s;
  }
}

// ---------------- 3. WMMA self-similarity at 64x64 --------------------------
// One wave per 4x4 center tile. Neighborhood union = 10x10 pixels (pad->112).
// Gram of A(16 centers x 64ch, f16) with B(16 neighbors x 64ch) via two
// v_wmma_f32_16x16x32_f16 (K=64 in two 32-steps).
__global__ __launch_bounds__(32)
void sim_wmma_kernel(const float* __restrict__ X, float* __restrict__ simOut) {
  const int tx = blockIdx.x, ty = blockIdx.y, b = blockIdx.z;
  const int lane = threadIdx.x;

  __shared__ unsigned int feat[112 * 32];  // [pixel][f16 channel-pair]
  __shared__ float nrm[112];

  const int gy0 = ty * 4 - 3, gx0 = tx * 4 - 3;
  const float* Xb = X + (size_t)b * 64 * 64 * 64;

  // Load 10x10 patch (zero for OOB / pad pixels), packed f16 pairs.
  for (int idx = lane; idx < 112 * 32; idx += 32) {
    int p = idx >> 5, j = idx & 31;
    unsigned int w = 0;
    if (p < 100) {
      int gy = gy0 + p / 10, gx = gx0 + p % 10;
      if (gy >= 0 && gy < 64 && gx >= 0 && gx < 64) {
        float c0 = Xb[((2 * j) * 64 + gy) * 64 + gx];
        float c1 = Xb[((2 * j + 1) * 64 + gy) * 64 + gx];
        union { _Float16 h[2]; unsigned int u; } pk;
        pk.h[0] = (_Float16)c0; pk.h[1] = (_Float16)c1;
        w = pk.u;
      }
    }
    feat[idx] = w;
  }
  __syncthreads();

  // Squared norms from the f16-rounded values (consistent with the Gram).
  for (int p = lane; p < 112; p += 32) {
    float s = 0.f;
    for (int j = 0; j < 32; ++j) {
      union { unsigned int u; _Float16 h[2]; } pk; pk.u = feat[p * 32 + j];
      float h0 = (float)pk.h[0], h1 = (float)pk.h[1];
      s += h0 * h0 + h1 * h1;
    }
    nrm[p] = s;
  }
  __syncthreads();

  const int Mrow = lane & 15;
  const int hi   = lane >> 4;                       // lane half
  const int cy0  = 3 + (Mrow >> 2), cx0 = 3 + (Mrow & 3);
  const int cIdxA = cy0 * 10 + cx0;

  union Frag { v16h h; unsigned int u[8]; };
  // A fragment (ISA 16-bit A 16x32 layout): VGPR v, pair = (v<4?0:8)+hi*4+(v&3)
  Frag a0, a1;
  #pragma unroll
  for (int v = 0; v < 8; ++v) {
    int pr = ((v < 4) ? 0 : 8) + (hi ? 4 : 0) + (v & 3);
    a0.u[v] = feat[cIdxA * 32 + pr];
    a1.u[v] = feat[cIdxA * 32 + pr + 16];
  }

  for (int t = 0; t < 7; ++t) {
    int n = t * 16 + (lane & 15);                   // neighbor column (padded)
    // B fragment (same K order as A): pair = (v<4?0:8)+hi*4+(v&3)
    Frag b0f, b1f;
    #pragma unroll
    for (int v = 0; v < 8; ++v) {
      int pr = ((v < 4) ? 0 : 8) + (hi ? 4 : 0) + (v & 3);
      b0f.u[v] = feat[n * 32 + pr];
      b1f.u[v] = feat[n * 32 + pr + 16];
    }
    v8f acc = {};
    acc = __builtin_amdgcn_wmma_f32_16x16x32_f16(false, a0.h, false, b0f.h,
                                                 (short)0, acc, false, false);
    acc = __builtin_amdgcn_wmma_f32_16x16x32_f16(false, a1.h, false, b1f.h,
                                                 (short)0, acc, false, false);

    // C/D layout: lane holds N = lane&15, M = r + 8*hi.
    if (n < 100) {
      int py = n / 10, px = n % 10;
      float nN = nrm[n];
      #pragma unroll
      for (int r = 0; r < 8; ++r) {
        int m  = r + 8 * hi;
        int cy = 3 + (m >> 2), cx = 3 + (m & 3);
        int dy = py - cy, dx = px - cx;
        if (dy >= -3 && dy <= 3 && dx >= -3 && dx <= 3) {
          int k49  = (dy + 3) * 7 + (dx + 3);
          float d2 = nrm[cy * 10 + cx] + nN - 2.0f * acc[r];
          float sv = __expf(-d2 * (1.0f / SIG2));
          int y = ty * 4 + (m >> 2), x = tx * 4 + (m & 3);
          simOut[(((size_t)b * K2 + k49) * 64 + y) * 64 + x] = sv;
        }
      }
    }
  }
}

// ---------------- 4. src_sim pos/neg masked sums ---------------------------
__global__ __launch_bounds__(256)
void srcmean_kernel(const int* __restrict__ gt, const float* __restrict__ simS,
                    float* __restrict__ part) {
  __shared__ float red[256];
  int i = blockIdx.x * 256 + threadIdx.x;
  int b = i >> 14, yx = i & (HWIM - 1);
  int y = yx >> 7, x = yx & 127;
  int g = gt[(size_t)b * HWIM + yx];
  int hy = y >> 1, hx = x >> 1;
  float sp = 0.f, np = 0.f, sn = 0.f, nn = 0.f;
  for (int k = 0; k < K2; ++k) {
    int dy = (k / 7 - 3) * 2, dx = (k % 7 - 3) * 2;
    int ny = y + dy, nx = x + dx;
    int gu = 0;                                     // zero pad -> label 0
    if (ny >= 0 && ny < 128 && nx >= 0 && nx < 128)
      gu = gt[(size_t)b * HWIM + ny * 128 + nx];
    float s = simS[(((size_t)b * K2 + k) * 64 + hy) * 64 + hx];
    if (gu == g) { sp += s; np += 1.f; } else { sn += s; nn += 1.f; }
  }
  float r0 = block_reduce(sp, red);
  float r1 = block_reduce(np, red);
  float r2 = block_reduce(sn, red);
  float r3 = block_reduce(nn, red);
  if (threadIdx.x == 0) {
    part[blockIdx.x * 4 + 0] = r0;
    part[blockIdx.x * 4 + 1] = r1;
    part[blockIdx.x * 4 + 2] = r2;
    part[blockIdx.x * 4 + 3] = r3;
  }
}

// ---------------- 5. top-9 / bottom-8 selection losses ---------------------
__global__ __launch_bounds__(256)
void topk_kernel(const float* __restrict__ simE, const float* __restrict__ cpp,
                 float* __restrict__ part) {
  __shared__ float red[256];
  int i = blockIdx.x * 256 + threadIdx.x;
  int b = i >> 14, yx = i & (HWIM - 1);
  int y = yx >> 7, x = yx & 127;
  int hy = y >> 1, hx = x >> 1;
  float s[K2];
  for (int k = 0; k < K2; ++k)
    s[k] = simE[(((size_t)b * K2 + k) * 64 + hy) * 64 + hx];

  unsigned long long used = 0ull;
  float accp = 0.f;
  for (int j = 0; j < 9; ++j) {                     // TOPK+1 largest
    float best = -1e30f; int bi = 0;
    for (int k = 0; k < K2; ++k)
      if (!((used >> k) & 1ull) && s[k] > best) { best = s[k]; bi = k; }
    used |= 1ull << bi;
    float c = cpp[(((size_t)b * K2 + bi) << 14) + yx];
    accp += best * (-c);
  }
  unsigned long long used2 = 0ull;
  float accn = 0.f;
  for (int j = 0; j < 8; ++j) {                     // TOPK smallest
    float worst = 1e30f; int wi = 0;
    for (int k = 0; k < K2; ++k)
      if (!((used2 >> k) & 1ull) && s[k] < worst) { worst = s[k]; wi = k; }
    used2 |= 1ull << wi;
    float c = cpp[(((size_t)b * K2 + wi) << 14) + yx];
    accn += (1.0f - worst) * (-(1.0f - c));
  }
  float r0 = block_reduce(accp, red);
  float r1 = block_reduce(accn, red);
  if (threadIdx.x == 0) {
    part[blockIdx.x * 2 + 0] = r0;
    part[blockIdx.x * 2 + 1] = r1;
  }
}

// ---------------- 6. final combine ----------------------------------------
__global__ void final_kernel(const float* __restrict__ p1,
                             const float* __restrict__ p2,
                             float* __restrict__ out) {
  if (threadIdx.x != 0 || blockIdx.x != 0) return;
  float sp = 0.f, np = 0.f, sn = 0.f, nn = 0.f, tp = 0.f, tn = 0.f;
  for (int i = 0; i < 128; ++i) {
    sp += p1[i * 4 + 0]; np += p1[i * 4 + 1];
    sn += p1[i * 4 + 2]; nn += p1[i * 4 + 3];
  }
  for (int i = 0; i < 128; ++i) { tp += p2[i * 2]; tn += p2[i * 2 + 1]; }
  out[0] = -(sp / np);                        // -src_pos_mean * W_SRC_POS
  out[1] = sn / nn;                           //  src_neg_mean * W_SRC_NEG
  out[2] = tp / (float)(NPIX * 9);            //  loss_sim_pos
  out[3] = tn / (float)(NPIX * 8);            //  loss_sim_neg
}

// ---------------------------------------------------------------------------
extern "C" void kernel_launch(void* const* d_in, const int* in_sizes, int n_in,
                              void* d_out, int out_size, void* d_ws, size_t ws_size,
                              hipStream_t stream) {
  (void)in_sizes; (void)n_in; (void)out_size; (void)ws_size;
  // setup_inputs order: logits_src, logits_trg, gt_src, img_trg, x_ema, x_src
  const float* logits_trg = (const float*)d_in[1];
  const int*   gt_src     = (const int*)d_in[2];
  const float* x_ema      = (const float*)d_in[4];
  const float* x_src      = (const float*)d_in[5];
  float* out = (float*)d_out;

  // workspace layout (floats); total ~12.1 MB
  float* ws   = (float*)d_ws;
  float* prob = ws;                         // 2*19*128*128 = 622592
  float* cpp  = prob + 622592;              // 2*49*128*128 = 1605632
  float* simE = cpp + 1605632;              // 2*49*64*64   = 401408
  float* simS = simE + 401408;              // 401408
  float* p1   = simS + 401408;              // 128*4
  float* p2   = p1 + 512;                   // 128*2

  softmax_kernel<<<NPIX / 256, 256, 0, stream>>>(logits_trg, prob);
  cpp_kernel<<<NPIX / 256, 256, 0, stream>>>(prob, cpp);
  sim_wmma_kernel<<<dim3(16, 16, 2), 32, 0, stream>>>(x_ema, simE);
  sim_wmma_kernel<<<dim3(16, 16, 2), 32, 0, stream>>>(x_src, simS);
  srcmean_kernel<<<NPIX / 256, 256, 0, stream>>>(gt_src, simS, p1);
  topk_kernel<<<NPIX / 256, 256, 0, stream>>>(simE, cpp, p2);
  final_kernel<<<1, 1, 0, stream>>>(p1, p2, out);
}
